// GRNN_7292854469118
// MI455X (gfx1250) — matrix operand order repaired and checked by reading, not compiled, not run on hardware
//
#include <hip/hip_runtime.h>

// ---------------------------------------------------------------------------
// GRNN fused kernel for gfx1250 (MI455X).
//   dist2[n,m] = |x_n|^2 + |y_m|^2 - 2 x_n.y_m      (bf16 WMMA cross GEMM)
//   w = exp(-dist2/2)                                (f32 VALU)
//   out[n,k] = (sum_m w*labels[m,k]) / (sum_m w)     (bf16 WMMA + atomics)
// ---------------------------------------------------------------------------

typedef __attribute__((ext_vector_type(16))) __bf16        v16bf;
typedef __attribute__((ext_vector_type(8)))  float         v8f;
typedef __attribute__((ext_vector_type(4)))  unsigned int  u32x4;

union FragBF { u32x4 u[2]; v16bf v; };

#define BR   128          // x rows per workgroup
#define BC   128          // train rows per c-block
#define CB   4            // c-blocks per workgroup (grid.y covers BC*CB = 512)
#define KT   64           // D-chunk staged in LDS per iteration
#define LDA  (KT + 8)     // padded LDS row stride (bf16) for A/B stages
#define LDW  (BC + 8)     // padded LDS row stride (bf16) for weight tile

// --- prep: f32 row -> bf16 row + squared norm ------------------------------
__global__ __launch_bounds__(256) void prep_rows(
    const float* __restrict__ src, __bf16* __restrict__ dst,
    float* __restrict__ sq, int D) {
  const int row = blockIdx.x;
  const float* r = src + (size_t)row * D;
  __bf16* o = dst + (size_t)row * D;
  float s = 0.f;
  for (int j = threadIdx.x; j < D; j += 256) {
    float v = r[j];
    s += v * v;
    o[j] = (__bf16)v;
  }
  __shared__ float red[256];
  red[threadIdx.x] = s;
  __syncthreads();
  for (int off = 128; off > 0; off >>= 1) {
    if (threadIdx.x < off) red[threadIdx.x] += red[threadIdx.x + off];
    __syncthreads();
  }
  if (threadIdx.x == 0) sq[row] = red[0];
}

// --- prep: labels [M,K] f32 -> [K,M] bf16 ----------------------------------
__global__ __launch_bounds__(256) void prep_labels_t(
    const float* __restrict__ tl, __bf16* __restrict__ lt, int M) {
  int idx = blockIdx.x * 256 + threadIdx.x;
  if (idx < M * 16) {
    int m = idx >> 4, k = idx & 15;
    lt[(size_t)k * M + m] = (__bf16)tl[idx];
  }
}

__global__ __launch_bounds__(256) void zero_f32(float* p, int n) {
  int i = blockIdx.x * 256 + threadIdx.x;
  if (i < n) p[i] = 0.f;
}

__global__ __launch_bounds__(256) void finalize_div(
    const float* __restrict__ accO, const float* __restrict__ accS,
    float* __restrict__ out, int n) {
  int i = blockIdx.x * 256 + threadIdx.x;
  if (i < n) out[i] = accO[i] / accS[i >> 4];
}

// --- main fused kernel ------------------------------------------------------
__global__ __launch_bounds__(256) void grnn_main(
    const __bf16* __restrict__ xbf,   // [N,512]
    const __bf16* __restrict__ ybf,   // [M,512]
    const __bf16* __restrict__ ltbf,  // [16,M]
    const float*  __restrict__ xsq,   // [N]
    const float*  __restrict__ ysq,   // [M]
    float* __restrict__ accO,         // [N,16]
    float* __restrict__ accS,         // [N]
    int Mtotal) {
  // stage buffers and the weight tile alias each other (phase-separated).
  __shared__ __bf16 smem[BR * LDA * 2];   // 36 KB
  __bf16* sA = smem;
  __bf16* sB = smem + BR * LDA;
  __bf16* sW = smem;                      // needs BR*LDW = 17408 <= 18432 elems

  const int tid  = threadIdx.x;
  const int w    = tid >> 5;        // wave id 0..7 -> owns x rows [16w,16w+16)
  const int lane = tid & 31;
  const int l16  = lane & 15;
  const int hi   = lane >> 4;
  const int rbase   = blockIdx.x * BR;
  const int r_local = 16 * w + l16;

  // squared norms for this wave's 16 rows in C-layout order (m = v + 8*hi)
  float xs[8];
#pragma unroll
  for (int v = 0; v < 8; ++v) xs[v] = xsq[rbase + 16 * w + v + 8 * hi];

  v8f   out_acc = {};        // persistent 16x16 (rows x K) label accumulator
  float wsum_acc[8] = {0.f, 0.f, 0.f, 0.f, 0.f, 0.f, 0.f, 0.f};

  for (int cb = 0; cb < CB; ++cb) {
    const int c0 = blockIdx.y * (BC * CB) + cb * BC;

    v8f acc[8];
#pragma unroll
    for (int t = 0; t < 8; ++t) acc[t] = (v8f){};

    // ---- cross GEMM over D=512 in KT=64 chunks ----
    for (int kt = 0; kt < 512; kt += KT) {
      // stage 128x64 bf16 tiles of x-block (A) and y-block (B)
#pragma unroll
      for (int i = 0; i < 4; ++i) {
        int u = tid + i * 256;          // 0..1023 : 128 rows x 8 (16B units)
        int r = u >> 3, kc = u & 7;
        u32x4 a = *(const u32x4*)(xbf + (((size_t)(rbase + r)) << 9) + kt + kc * 8);
        u32x4 b = *(const u32x4*)(ybf + (((size_t)(c0 + r)) << 9) + kt + kc * 8);
        *(u32x4*)(sA + r * LDA + kc * 8) = a;
        *(u32x4*)(sB + r * LDA + kc * 8) = b;
      }
      if (kt + KT < 512) {   // warm L2/L0 for the next chunk
        __builtin_prefetch(xbf + (((size_t)(rbase + (tid >> 1))) << 9) + kt + KT, 0, 1);
        __builtin_prefetch(ybf + (((size_t)(c0 + (tid >> 1))) << 9) + kt + KT, 0, 1);
      }
      __syncthreads();

#pragma unroll
      for (int kk = 0; kk < KT; kk += 32) {
        // A fragment: 16x32 bf16, lane<16 holds k {0..7,16..23}, lane>=16 {8..15,24..31}
        FragBF a;
        const __bf16* ap = sA + r_local * LDA + kk + 8 * hi;
        a.u[0] = *(const u32x4*)(ap);
        a.u[1] = *(const u32x4*)(ap + 16);
#pragma unroll
        for (int ct = 0; ct < 8; ++ct) {
          // B fragment: col n = l16, lane<16 holds k {0..7,16..23}, lane>=16 {8..15,24..31}
          FragBF b;
          const __bf16* bp = sB + (16 * ct + l16) * LDA + kk + 8 * hi;
          b.u[0] = *(const u32x4*)(bp);
          b.u[1] = *(const u32x4*)(bp + 16);
          acc[ct] = __builtin_amdgcn_wmma_f32_16x16x32_bf16(
              false, a.v, false, b.v, (short)0, acc[ct], false, false);
        }
      }
      __syncthreads();
    }

    // ---- epilogue: dist -> exp, weight tile to LDS (bf16), row sums ----
    float rowpart[8] = {0.f, 0.f, 0.f, 0.f, 0.f, 0.f, 0.f, 0.f};
#pragma unroll
    for (int ct = 0; ct < 8; ++ct) {
      const float ys = ysq[c0 + 16 * ct + l16];
#pragma unroll
      for (int v = 0; v < 8; ++v) {
        float d   = xs[v] + ys - 2.0f * acc[ct][v];
        float wgt = __expf(-0.5f * d);          // SPREAD = 1.0
        rowpart[v] += wgt;
        sW[(16 * w + v + 8 * hi) * LDW + 16 * ct + l16] = (__bf16)wgt;
      }
    }
    // reduce weight row-sums across the 16-lane half-wave
#pragma unroll
    for (int v = 0; v < 8; ++v) {
      float s = rowpart[v];
      s += __shfl_xor(s, 1, 32);
      s += __shfl_xor(s, 2, 32);
      s += __shfl_xor(s, 4, 32);
      s += __shfl_xor(s, 8, 32);
      wsum_acc[v] += s;     // valid on lanes with l16 == 0
    }

    // ---- second WMMA: out_acc += W(16x128) * labelsT-block(128x16) ----
#pragma unroll
    for (int kk = 0; kk < BC; kk += 32) {
      FragBF a, b;
      const __bf16* ap = sW + r_local * LDW + kk + 8 * hi;
      a.u[0] = *(const u32x4*)(ap);
      a.u[1] = *(const u32x4*)(ap + 16);
      const __bf16* bp = ltbf + (size_t)l16 * Mtotal + c0 + kk + 8 * hi;
      b.u[0] = *(const u32x4*)(bp);
      b.u[1] = *(const u32x4*)(bp + 16);
      out_acc = __builtin_amdgcn_wmma_f32_16x16x32_bf16(
          false, a.v, false, b.v, (short)0, out_acc, false, false);
    }
    __syncthreads();   // before next c-block re-stages over sW alias
  }

  // ---- combine M-split partials ----
#pragma unroll
  for (int v = 0; v < 8; ++v) {
    int row = rbase + 16 * w + v + 8 * hi;
    atomicAdd(&accO[(size_t)row * 16 + l16], out_acc[v]);
  }
  if (l16 == 0) {
#pragma unroll
    for (int v = 0; v < 8; ++v) {
      int row = rbase + 16 * w + v + 8 * hi;
      atomicAdd(&accS[row], wsum_acc[v]);
    }
  }
}

// ---------------------------------------------------------------------------
extern "C" void kernel_launch(void* const* d_in, const int* in_sizes, int n_in,
                              void* d_out, int out_size, void* d_ws, size_t ws_size,
                              hipStream_t stream) {
  const float* x  = (const float*)d_in[0];   // [N,512]
  const float* td = (const float*)d_in[1];   // [M,512]
  const float* tl = (const float*)d_in[2];   // [M,16]
  const int D = 512, K = 16;
  const int N = in_sizes[0] / D;             // 4096
  const int M = in_sizes[1] / D;             // 8192

  char* ws = (char*)d_ws;
  size_t off = 0;
  auto alloc = [&](size_t bytes) -> void* {
    void* p = ws + off;
    off += (bytes + 255) & ~(size_t)255;
    return p;
  };
  __bf16* xbf  = (__bf16*)alloc((size_t)N * D * 2);
  __bf16* ybf  = (__bf16*)alloc((size_t)M * D * 2);
  __bf16* ltbf = (__bf16*)alloc((size_t)K * M * 2);
  float*  xsq  = (float*)alloc((size_t)N * 4);
  float*  ysq  = (float*)alloc((size_t)M * 4);
  float*  accO = (float*)alloc((size_t)N * K * 4);
  float*  accS = (float*)alloc((size_t)N * 4);

  prep_rows<<<N, 256, 0, stream>>>(x, xbf, xsq, D);
  prep_rows<<<M, 256, 0, stream>>>(td, ybf, ysq, D);
  prep_labels_t<<<(M * K + 255) / 256, 256, 0, stream>>>(tl, ltbf, M);
  zero_f32<<<(N * K + 255) / 256, 256, 0, stream>>>(accO, N * K);
  zero_f32<<<(N + 255) / 256, 256, 0, stream>>>(accS, N);

  dim3 grid(N / BR, M / (BC * CB));   // 32 x 16 = 512 workgroups
  grnn_main<<<grid, 256, 0, stream>>>(xbf, ybf, ltbf, xsq, ysq, accO, accS, M);

  finalize_div<<<(N * K + 255) / 256, 256, 0, stream>>>(accO, accS,
                                                        (float*)d_out, N * K);
}
